// GraphDecoder_41248865911346
// MI455X (gfx1250) — matrix operand order repaired and checked by reading, not compiled, not run on hardware
//
#include <hip/hip_runtime.h>
#include <hip/hip_bf16.h>

typedef __attribute__((ext_vector_type(16))) _Float16 v16h;
typedef __attribute__((ext_vector_type(8)))  _Float16 v8h;
typedef __attribute__((ext_vector_type(8)))  float    v8f;

#define TB 256

// ---------------------------------------------------------------- utilities
__global__ void zero_kernel(float* __restrict__ p, int n) {
    int i = blockIdx.x * blockDim.x + threadIdx.x;
    if (i < n) p[i] = 0.0f;
}

__global__ void deg_kernel(const int* __restrict__ dst, float* __restrict__ cnt, int E) {
    int e = blockIdx.x * blockDim.x + threadIdx.x;
    if (e < E) atomicAdd(&cnt[dst[e]], 1.0f);
}

__global__ void inv_kernel(const float* __restrict__ cnt, float* __restrict__ inv, int N) {
    int i = blockIdx.x * blockDim.x + threadIdx.x;
    if (i < N) inv[i] = 1.0f / fmaxf(cnt[i], 1.0f);
}

// Pack [Wl;Wr] (each Fin x Fout, row-major f32) into BT[n][k] f16, k in [0,2*Fin)
__global__ void packBT_kernel(const float* __restrict__ Wl, const float* __restrict__ Wr,
                              _Float16* __restrict__ BT, int Fin, int Fout) {
    int gid = blockIdx.x * blockDim.x + threadIdx.x;
    int K2 = 2 * Fin;
    if (gid >= Fout * K2) return;
    int n = gid / K2, k = gid % K2;
    float w = (k < Fin) ? Wl[k * Fout + n] : Wr[(k - Fin) * Fout + n];
    BT[(size_t)n * K2 + k] = (_Float16)w;
}

// ------------------------------------------------------------- scatter adds
__global__ void scatter3_kernel(const float* __restrict__ z, const int* __restrict__ src,
                                const int* __restrict__ dst, float* __restrict__ agg, int E) {
    int e = blockIdx.x * blockDim.x + threadIdx.x;
    if (e >= E) return;
    int s = src[e], d = dst[e];
    atomicAdd(&agg[d * 3 + 0], z[s * 3 + 0]);
    atomicAdd(&agg[d * 3 + 1], z[s * 3 + 1]);
    atomicAdd(&agg[d * 3 + 2], z[s * 3 + 2]);
}

// F = 128: 32 threads/edge, 4 floats/thread (coalesced row gather, L2 atomics)
__global__ void scatter128_kernel(const float* __restrict__ X, const int* __restrict__ src,
                                  const int* __restrict__ dst, float* __restrict__ agg, int E) {
    int gid = blockIdx.x * blockDim.x + threadIdx.x;
    int e  = gid >> 5;
    int fi = (gid & 31) * 4;
    if (e >= E) return;
    int s = src[e], d = dst[e];
    const float4 v = *(const float4*)(X + (size_t)s * 128 + fi);
    float* a = agg + (size_t)d * 128 + fi;
    atomicAdd(a + 0, v.x); atomicAdd(a + 1, v.y);
    atomicAdd(a + 2, v.z); atomicAdd(a + 3, v.w);
}

// mean -> f16 into front half of [mean|x] operand (row stride 256 halves)
__global__ void prep_mean_kernel(const float* __restrict__ agg, const float* __restrict__ inv,
                                 _Float16* __restrict__ Ah, int N) {
    int gid = blockIdx.x * blockDim.x + threadIdx.x;
    if (gid >= N * 128) return;
    int node = gid >> 7, f = gid & 127;
    Ah[(size_t)node * 256 + f] = (_Float16)(agg[gid] * inv[node]);
}

// ------------------------------------------------------------------ layer 1
__global__ void layer1_kernel(const float* __restrict__ z, const float* __restrict__ agg,
                              const float* __restrict__ inv,
                              const float* __restrict__ W1l, const float* __restrict__ W1r,
                              const float* __restrict__ b1,
                              float* __restrict__ X, _Float16* __restrict__ A2h, int N) {
    int gid = blockIdx.x * blockDim.x + threadIdx.x;
    if (gid >= N * 128) return;
    int node = gid >> 7, f = gid & 127;
    float iv = inv[node];
    float m0 = agg[node * 3 + 0] * iv, m1 = agg[node * 3 + 1] * iv, m2 = agg[node * 3 + 2] * iv;
    float x0 = z[node * 3 + 0], x1 = z[node * 3 + 1], x2 = z[node * 3 + 2];
    float v = b1[f];
    v = fmaf(m0, W1l[f], v); v = fmaf(m1, W1l[128 + f], v); v = fmaf(m2, W1l[256 + f], v);
    v = fmaf(x0, W1r[f], v); v = fmaf(x1, W1r[128 + f], v); v = fmaf(x2, W1r[256 + f], v);
    v = fmaxf(v, 0.0f);
    X[(size_t)node * 128 + f] = v;
    A2h[(size_t)node * 256 + 128 + f] = (_Float16)v;
}

// ------------------------------------------------- WMMA GEMM (layers 2 & 3)
// D[16 x FOUT] = relu(A[16 x K] * B[K x FOUT] + bias); A f16 row-major,
// BT f16 = B transposed [FOUT x K]; 4 waves/block, FOUT/64 n-tiles per wave.
template <int K, int FOUT, bool WRITE_HALF>
__launch_bounds__(128)
__global__ void sage_gemm_kernel(const _Float16* __restrict__ A,
                                 const _Float16* __restrict__ BT,
                                 const float* __restrict__ bias,
                                 float* __restrict__ Xout,
                                 _Float16* __restrict__ Hout,
                                 int hStride, int hOff, int nrows) {
    constexpr int TPW = FOUT / 64;            // 16-wide n-tiles per wave
    const int wave   = threadIdx.x >> 5;
    const int lane   = threadIdx.x & 31;
    const int half16 = (lane < 16) ? 0 : 1;
    const int lane15 = lane & 15;
    const int row0   = blockIdx.x << 4;

    int rA = row0 + lane15; if (rA > nrows - 1) rA = nrows - 1;
    const _Float16* aRow = A + (size_t)rA * K + half16 * 8;
    const int nBase = wave * (TPW * 16);

    v8f acc[TPW] = {};

    #pragma unroll 1
    for (int k0 = 0; k0 < K; k0 += 32) {
        // A fragment: lane<16 -> K {k0..+7, k0+16..+23}; lane>=16 -> {+8..+15, +24..+31}
        v8h alo = *(const v8h*)(aRow + k0);
        v8h ahi = *(const v8h*)(aRow + k0 + 16);
        v16h afrag;
        #pragma unroll
        for (int i = 0; i < 8; ++i) { afrag[i] = alo[i]; afrag[i + 8] = ahi[i]; }

        #pragma unroll
        for (int t = 0; t < TPW; ++t) {
            // B fragment: lane<16 -> col nBase+t*16+lane, K {k0..+7, k0+16..+23}; lane>=16 -> {+8..+15, +24..+31}
            const _Float16* bp = BT + (size_t)(nBase + t * 16 + lane15) * K + k0 + half16 * 8;
            v8h blo = *(const v8h*)(bp);
            v8h bhi = *(const v8h*)(bp + 16);
            v16h bfrag;
            #pragma unroll
            for (int i = 0; i < 8; ++i) { bfrag[i] = blo[i]; bfrag[i + 8] = bhi[i]; }

            acc[t] = __builtin_amdgcn_wmma_f32_16x16x32_f16(
                false, afrag, false, bfrag, (short)0, acc[t], false, false);
        }
    }

    // Epilogue: VGPR v -> M = v + 8*half16, N = lane15 (ISA 7.12.2 C/D layout).
    // Uniform full-tile test hoists the row guard out of the store sequence:
    // at N % 16 == 0 every block takes the branch-free path.
    const int rowB = row0 + half16 * 8;
    if (row0 + 16 <= nrows) {
        #pragma unroll
        for (int t = 0; t < TPW; ++t) {
            const int col = nBase + t * 16 + lane15;
            const float bcol = bias[col];
            #pragma unroll
            for (int v = 0; v < 8; ++v) {
                const int row = rowB + v;
                float val = fmaxf(acc[t][v] + bcol, 0.0f);
                Xout[(size_t)row * FOUT + col] = val;
                if (WRITE_HALF)
                    Hout[(size_t)row * hStride + hOff + col] = (_Float16)val;
            }
        }
    } else {
        #pragma unroll
        for (int t = 0; t < TPW; ++t) {
            const int col = nBase + t * 16 + lane15;
            const float bcol = bias[col];
            #pragma unroll
            for (int v = 0; v < 8; ++v) {
                const int row = rowB + v;
                if (row < nrows) {
                    float val = fmaxf(acc[t][v] + bcol, 0.0f);
                    Xout[(size_t)row * FOUT + col] = val;
                    if (WRITE_HALF)
                        Hout[(size_t)row * hStride + hOff + col] = (_Float16)val;
                }
            }
        }
    }
}

// ------------------------------------------------------------------ layer 4
__global__ void layer4_kernel(const float* __restrict__ X3, const float* __restrict__ W4,
                              const float* __restrict__ b4, float* __restrict__ out, int N) {
    int gid = blockIdx.x * blockDim.x + threadIdx.x;
    if (gid >= N * 40) return;
    int node = gid / 40, f = gid % 40;
    const float* x = X3 + (size_t)node * 64;
    float acc = b4[f];
    #pragma unroll 8
    for (int k = 0; k < 64; ++k) acc = fmaf(x[k], W4[k * 40 + f], acc);
    out[gid] = acc;
}

// --------------------------------------------------------------------- host
extern "C" void kernel_launch(void* const* d_in, const int* in_sizes, int n_in,
                              void* d_out, int out_size, void* d_ws, size_t ws_size,
                              hipStream_t stream) {
    const float* z   = (const float*)d_in[0];
    const int*   ei  = (const int*)d_in[1];
    const float* W1l = (const float*)d_in[2];
    const float* W1r = (const float*)d_in[3];
    const float* b1  = (const float*)d_in[4];
    const float* W2l = (const float*)d_in[5];
    const float* W2r = (const float*)d_in[6];
    const float* b2  = (const float*)d_in[7];
    const float* W3l = (const float*)d_in[8];
    const float* W3r = (const float*)d_in[9];
    const float* b3  = (const float*)d_in[10];
    const float* W4  = (const float*)d_in[11];
    const float* b4  = (const float*)d_in[12];

    const int N = in_sizes[0] / 3;
    const int E = in_sizes[1] / 2;
    const int* src = ei;
    const int* dst = ei + E;

    // workspace carve-out (256B aligned)
    char* ws = (char*)d_ws;
    size_t off = 0;
    auto alloc = [&](size_t bytes) -> void* {
        void* p = ws + off;
        off += (bytes + 255) & ~(size_t)255;
        return p;
    };
    float*    cnt = (float*)alloc((size_t)N * 4);
    float*    inv = (float*)alloc((size_t)N * 4);
    float*    agg = (float*)alloc((size_t)N * 128 * 4);   // reused each layer
    float*    X   = (float*)alloc((size_t)N * 128 * 4);   // reused activations
    _Float16* A2h = (_Float16*)alloc((size_t)N * 256 * 2);
    _Float16* A3h = (_Float16*)alloc((size_t)N * 256 * 2);
    _Float16* B2T = (_Float16*)alloc((size_t)128 * 256 * 2);
    _Float16* B3T = (_Float16*)alloc((size_t)64 * 256 * 2);

    auto gridFor = [](long long n) { return (int)((n + TB - 1) / TB); };

    // pack weights once per launch (tiny)
    packBT_kernel<<<gridFor(128 * 256), TB, 0, stream>>>(W2l, W2r, B2T, 128, 128);
    packBT_kernel<<<gridFor(64 * 256),  TB, 0, stream>>>(W3l, W3r, B3T, 128, 64);

    // degrees
    zero_kernel<<<gridFor(N), TB, 0, stream>>>(cnt, N);
    deg_kernel<<<gridFor(E), TB, 0, stream>>>(dst, cnt, E);
    inv_kernel<<<gridFor(N), TB, 0, stream>>>(cnt, inv, N);

    // ---- layer 1 (K=6, scalar) ----
    zero_kernel<<<gridFor((long long)N * 3), TB, 0, stream>>>(agg, N * 3);
    scatter3_kernel<<<gridFor(E), TB, 0, stream>>>(z, src, dst, agg, E);
    layer1_kernel<<<gridFor((long long)N * 128), TB, 0, stream>>>(z, agg, inv, W1l, W1r, b1,
                                                                  X, A2h, N);

    // ---- layer 2 (WMMA, K=256 -> 128) ----
    zero_kernel<<<gridFor((long long)N * 128), TB, 0, stream>>>(agg, N * 128);
    scatter128_kernel<<<gridFor((long long)E * 32), TB, 0, stream>>>(X, src, dst, agg, E);
    prep_mean_kernel<<<gridFor((long long)N * 128), TB, 0, stream>>>(agg, inv, A2h, N);
    sage_gemm_kernel<256, 128, true><<<(N + 15) / 16, 128, 0, stream>>>(
        A2h, B2T, b2, X, A3h, 256, 128, N);

    // ---- layer 3 (WMMA, K=256 -> 64) ----
    zero_kernel<<<gridFor((long long)N * 128), TB, 0, stream>>>(agg, N * 128);
    scatter128_kernel<<<gridFor((long long)E * 32), TB, 0, stream>>>(X, src, dst, agg, E);
    prep_mean_kernel<<<gridFor((long long)N * 128), TB, 0, stream>>>(agg, inv, A3h, N);
    sage_gemm_kernel<256, 64, false><<<(N + 15) / 16, 128, 0, stream>>>(
        A3h, B3T, b3, X, (_Float16*)nullptr, 0, 0, N);

    // ---- layer 4 (64 -> 40, scalar) ----
    layer4_kernel<<<gridFor((long long)N * 40), TB, 0, stream>>>(X, W4, b4, (float*)d_out, N);
}
